// MultiHeadAttention_66365834658078
// MI455X (gfx1250) — matrix operand order repaired and checked by reading, not compiled, not run on hardware
//
#include <hip/hip_runtime.h>
#include <stdint.h>

#define Bb 4
#define Ss 2048
#define Dd 1024
#define Hh 16
#define Dh 64
#define Mrows (Bb*Ss)
#define NEGV -1000000000.0f

typedef __attribute__((ext_vector_type(16))) __bf16        v16bf;
typedef __attribute__((ext_vector_type(8)))  float         v8f;
typedef __attribute__((ext_vector_type(4)))  unsigned int  uvec4;
typedef __attribute__((ext_vector_type(4)))  float         fvec4;
typedef __attribute__((ext_vector_type(4)))  unsigned short usvec4;
typedef __attribute__((ext_vector_type(4)))  unsigned int  u32x4;
typedef __attribute__((ext_vector_type(8)))  int           i32x8;
typedef __attribute__((ext_vector_type(4)))  int           i32x4;

struct B32x2 { uvec4 lo, hi; };

__device__ inline unsigned short f2bf(float f) {
  unsigned int u = __float_as_uint(f);
  u += 0x7FFFu + ((u >> 16) & 1u);          // round-to-nearest-even
  return (unsigned short)(u >> 16);
}

// A-matrix 16x32 bf16 fragment (ISA 7.12.2): lane m = lane&15,
// K chunks of 8 at kb=(lane>>4)*8 and kb+16.
__device__ inline v16bf frag_a(const unsigned short* base, int ld, int lane) {
  const unsigned short* p = base + (lane & 15) * ld + ((lane >> 4) << 3);
  B32x2 t;
  t.lo = *(const uvec4*)(p);
  t.hi = *(const uvec4*)(p + 16);
  return __builtin_bit_cast(v16bf, t);
}

// B-matrix 32x16 bf16 fragment: lane n = lane&15, K chunks of 8 at kb=(lane>>4)*8 and kb+16.
__device__ inline v16bf frag_b(const unsigned short* base, int ld, int lane) {
  const unsigned short* p = base + (lane & 15) * ld + ((lane >> 4) << 3);
  B32x2 t;
  t.lo = *(const uvec4*)(p);
  t.hi = *(const uvec4*)(p + 16);
  return __builtin_bit_cast(v16bf, t);
}

__device__ inline v8f wmma_bf16(v16bf a, v16bf b, v8f c) {
  return __builtin_amdgcn_wmma_f32_16x16x32_bf16(false, a, false, b,
                                                 (short)0, c, false, false);
}

// ---------------------------------------------------------------------------
// TDM: 2D bf16 tile (tile_d0 elems/row x tile_d1 rows) global -> LDS,
// with LDS padding of 4 DWORDs (8 shorts) after every 32 DWORDs (64 shorts),
// i.e. LDS row stride 72 shorts for 64-element rows.
// D# packing per CDNA5 ISA 8.3/8.4 (group0 128b, group1 256b).
// 6-arg builtin form (clang-23 / therock headers).
// ---------------------------------------------------------------------------
__device__ inline void tdm_load_2d_bf16(unsigned int lds_byte_off,
                                        const void* gptr,
                                        unsigned int tile_d0, unsigned int tile_d1,
                                        unsigned int tensor_d0, unsigned int tensor_d1,
                                        unsigned int stride_elems) {
  unsigned long long ga = (unsigned long long)(uintptr_t)gptr;
  u32x4 g0;
  g0[0] = 1u;                                            // count=1 (valid), user mode
  g0[1] = lds_byte_off;                                  // lds_addr
  g0[2] = (unsigned int)(ga & 0xFFFFFFFFu);              // global_addr[31:0]
  g0[3] = (unsigned int)((ga >> 32) & 0x1FFFFFFu) | (2u << 30);  // [56:32] | type=2
  i32x8 g1;
  // data_size=1 (2 bytes), pad_enable, pad_interval code 4 (=32 DW), pad_amount code 3 (=4 DW)
  g1[0] = (int)((1u << 16) | (1u << 20) | (4u << 22) | (3u << 25));
  g1[1] = (int)((tensor_d0 & 0xFFFFu) << 16);            // atomic_barrier_addr=0 | td0[15:0]
  g1[2] = (int)(((tensor_d0 >> 16) & 0xFFFFu) | ((tensor_d1 & 0xFFFFu) << 16));
  g1[3] = (int)(((tensor_d1 >> 16) & 0xFFFFu) | ((tile_d0 & 0xFFFFu) << 16));
  g1[4] = (int)(tile_d1 & 0xFFFFu);                      // tile_dim1 | tile_dim2=0 (2D)
  g1[5] = (int)stride_elems;                             // tensor_dim0_stride[31:0]
  g1[6] = 0;
  g1[7] = 0;
  i32x4 z4 = { 0, 0, 0, 0 };
  i32x8 z8 = { 0, 0, 0, 0, 0, 0, 0, 0 };
  __builtin_amdgcn_tensor_load_to_lds(g0, g1, z4, z4, z8, 0);
}

// ---------------------------------------------------------------------------
// GEMM 1: X[8192,1024](f32) @ W[1024,1024](f32) + bias -> bf16 [B,H,S,64]
// 256 threads = 8 waves; tile 128x128x32; wave = 32x64 (2x4 WMMA tiles)
// ---------------------------------------------------------------------------
__global__ __launch_bounds__(256) void qkv_proj_kernel(
    const float* __restrict__ A, const float* __restrict__ W,
    const float* __restrict__ bias, unsigned short* __restrict__ outp)
{
  __shared__ unsigned short As[128 * 40];   // [m][k] k-contiguous
  __shared__ unsigned short Bsh[128 * 40];  // [n][k] k-contiguous (W transposed)
  const int tid = threadIdx.x, lane = tid & 31, wid = tid >> 5;
  const int wm = wid & 3, wn = wid >> 2;
  const int m0 = blockIdx.y * 128, n0 = blockIdx.x * 128;

  v8f acc[2][4];
  for (int i = 0; i < 2; ++i) for (int j = 0; j < 4; ++j) acc[i][j] = v8f{};

  const int tkA = (tid & 7) * 4, tmA = tid >> 3;     // A: 4 rows x float4
  const int tnB = (tid & 31) * 4, kgB = tid >> 5;    // B: 4 k-rows x float4

  for (int k0 = 0; k0 < Dd; k0 += 32) {
    for (int i = 0; i < 4; ++i) {
      int row = tmA + 32 * i;
      fvec4 av = *(const fvec4*)(A + (size_t)(m0 + row) * Dd + k0 + tkA);
      usvec4 sv = { f2bf(av[0]), f2bf(av[1]), f2bf(av[2]), f2bf(av[3]) };
      *(usvec4*)(&As[row * 40 + tkA]) = sv;
    }
    for (int j = 0; j < 4; ++j) {
      int k = kgB * 4 + j;
      fvec4 wv = *(const fvec4*)(W + (size_t)(k0 + k) * Dd + n0 + tnB);
      for (int c = 0; c < 4; ++c) Bsh[(tnB + c) * 40 + k] = f2bf(wv[c]);
    }
    if (k0 + 32 < Dd) {   // prefetch next k-step tiles -> global_prefetch_b8
      __builtin_prefetch((const void*)(A + (size_t)(m0 + tmA) * Dd + k0 + 32 + tkA), 0, 3);
      __builtin_prefetch((const void*)(W + (size_t)(k0 + 32 + kgB * 4) * Dd + n0 + tnB), 0, 3);
    }
    __syncthreads();
    v16bf a0 = frag_a(&As[(wm * 32) * 40], 40, lane);
    v16bf a1 = frag_a(&As[(wm * 32 + 16) * 40], 40, lane);
    for (int nt = 0; nt < 4; ++nt) {
      v16bf bq = frag_b(&Bsh[(wn * 64 + nt * 16) * 40], 40, lane);
      acc[0][nt] = wmma_bf16(a0, bq, acc[0][nt]);
      acc[1][nt] = wmma_bf16(a1, bq, acc[1][nt]);
    }
    __syncthreads();
  }

  const int mb = (lane >> 4) * 8, nn = lane & 15;
  for (int mt = 0; mt < 2; ++mt)
    for (int nt = 0; nt < 4; ++nt)
      for (int r = 0; r < 8; ++r) {
        int mg = m0 + wm * 32 + mt * 16 + mb + r;
        int ng = n0 + wn * 64 + nt * 16 + nn;
        float v = acc[mt][nt][r] + bias[ng];
        int b = mg >> 11, s = mg & (Ss - 1);
        int h = ng >> 6, dd = ng & 63;
        outp[(((size_t)(b * Hh + h) * Ss) + s) * Dh + dd] = f2bf(v);
      }
}

// ---------------------------------------------------------------------------
// Flash attention: 1 WG = (b,h) x 128-query block; 4 waves x 32 rows.
// Key blocks of 64; online softmax; causal mask analytic.
// Q and K tiles staged into LDS via Tensor Data Mover; V transposed manually.
// ---------------------------------------------------------------------------
__global__ __launch_bounds__(128) void attn_kernel(
    const unsigned short* __restrict__ qh, const unsigned short* __restrict__ kh,
    const unsigned short* __restrict__ vh, unsigned short* __restrict__ ctx)
{
  __shared__ unsigned short Qs[128 * 72];      // [query][d]   (TDM-padded rows)
  __shared__ unsigned short Ksh[64 * 72];      // [key][d]     (TDM-padded rows)
  __shared__ unsigned short Vt[64 * 72];       // [d][key] (transposed)
  __shared__ unsigned short Ps[4][32 * 72];    // per-wave P [row][key]

  const int tid = threadIdx.x, lane = tid & 31, wid = tid >> 5;
  const int qb = blockIdx.x, bh = blockIdx.y;
  const int b = bh / Hh, h = bh % Hh;
  const int q0 = qb * 128;

  // Q block via TDM (wave 0 issues; tensor = [S, 64] bf16 per head)
  if (wid == 0) {
    tdm_load_2d_bf16((unsigned int)(uintptr_t)(void*)&Qs[0],
                     qh + ((size_t)bh * Ss + q0) * Dh,
                     Dh, 128, Dh, Ss, Dh);
    __builtin_amdgcn_s_wait_tensorcnt(0);
  }

  v8f acc_o[2][4];
  for (int i = 0; i < 2; ++i) for (int j = 0; j < 4; ++j) acc_o[i][j] = v8f{};
  float m_i[2][8], l_i[2][8];
  for (int i = 0; i < 2; ++i) for (int r = 0; r < 8; ++r) { m_i[i][r] = -3.0e38f; l_i[i][r] = 0.0f; }

  const int mbase = (lane >> 4) * 8, nn = lane & 15;
  const int rowbase = wid * 32;
  const int jmax = 2 * qb + 2;   // causal: keys up to q0+127

  for (int jb = 0; jb < jmax; ++jb) {
    const int ks0 = jb * 64;
    __syncthreads();
    // K tile via TDM (wave 0)
    if (wid == 0) {
      tdm_load_2d_bf16((unsigned int)(uintptr_t)(void*)&Ksh[0],
                       kh + ((size_t)bh * Ss + ks0) * Dh,
                       Dh, 64, Dh, Ss, Dh);
    }
    // V tile transposed (all threads)
    {
      int key = tid >> 1, half = tid & 1;
      const unsigned short* vp = vh + ((size_t)bh * Ss + ks0 + key) * Dh + half * 32;
      for (int i = 0; i < 32; ++i)
        Vt[(half * 32 + i) * 72 + key] = vp[i];
    }
    if (wid == 0) __builtin_amdgcn_s_wait_tensorcnt(0);
    __syncthreads();

    // S = Q @ K^T  (K-dim = 64 -> two 32-wide WMMA steps)
    v8f sacc[2][4];
    for (int i = 0; i < 2; ++i) for (int j = 0; j < 4; ++j) sacc[i][j] = v8f{};
    for (int kk = 0; kk < 2; ++kk) {
      v16bf a0 = frag_a(&Qs[rowbase * 72 + kk * 32], 72, lane);
      v16bf a1 = frag_a(&Qs[(rowbase + 16) * 72 + kk * 32], 72, lane);
      for (int nt = 0; nt < 4; ++nt) {
        v16bf bk = frag_b(&Ksh[(nt * 16) * 72 + kk * 32], 72, lane);
        sacc[0][nt] = wmma_bf16(a0, bk, sacc[0][nt]);
        sacc[1][nt] = wmma_bf16(a1, bk, sacc[1][nt]);
      }
    }

    // online softmax (scale = 1/sqrt(64))
    for (int mt = 0; mt < 2; ++mt) {
      for (int r = 0; r < 8; ++r) {
        int rowg = q0 + rowbase + mt * 16 + mbase + r;
        float mnew = m_i[mt][r];
        for (int nt = 0; nt < 4; ++nt) {
          int colg = ks0 + nt * 16 + nn;
          float s = sacc[mt][nt][r] * 0.125f;
          if (colg > rowg) s = NEGV;
          sacc[mt][nt][r] = s;
          mnew = fmaxf(mnew, s);
        }
        for (int off = 1; off < 16; off <<= 1)
          mnew = fmaxf(mnew, __shfl_xor(mnew, off, 32));
        float alpha = __expf(m_i[mt][r] - mnew);
        m_i[mt][r] = mnew;
        float rs = 0.0f;
        for (int nt = 0; nt < 4; ++nt) {
          float p = __expf(sacc[mt][nt][r] - mnew);
          sacc[mt][nt][r] = p;
          rs += p;
        }
        for (int off = 1; off < 16; off <<= 1)
          rs += __shfl_xor(rs, off, 32);
        l_i[mt][r] = l_i[mt][r] * alpha + rs;
        for (int nt = 0; nt < 4; ++nt)
          acc_o[mt][nt][r] *= alpha;
      }
    }

    // P -> per-wave LDS (row-major, key contiguous), then O += P @ V
    unsigned short* pw = &Ps[wid][0];
    for (int mt = 0; mt < 2; ++mt)
      for (int nt = 0; nt < 4; ++nt)
        for (int r = 0; r < 8; ++r)
          pw[(mt * 16 + mbase + r) * 72 + nt * 16 + nn] = f2bf(sacc[mt][nt][r]);

    for (int kk = 0; kk < 2; ++kk) {
      v16bf a0 = frag_a(&pw[0 * 72 + kk * 32], 72, lane);
      v16bf a1 = frag_a(&pw[16 * 72 + kk * 32], 72, lane);
      for (int nt = 0; nt < 4; ++nt) {
        v16bf bv = frag_b(&Vt[(nt * 16) * 72 + kk * 32], 72, lane);
        acc_o[0][nt] = wmma_bf16(a0, bv, acc_o[0][nt]);
        acc_o[1][nt] = wmma_bf16(a1, bv, acc_o[1][nt]);
      }
    }
  }

  // O /= l ; write ctx [B,S,D] bf16, D index = h*64+dd
  for (int mt = 0; mt < 2; ++mt)
    for (int r = 0; r < 8; ++r) {
      int sg = q0 + rowbase + mt * 16 + mbase + r;
      float rinv = 1.0f / l_i[mt][r];
      for (int nt = 0; nt < 4; ++nt) {
        int dd = nt * 16 + nn;
        ctx[((size_t)(b * Ss + sg)) * Dd + h * Dh + dd] = f2bf(acc_o[mt][nt][r] * rinv);
      }
    }
}

// ---------------------------------------------------------------------------
// GEMM 2: ctx[8192,1024](bf16) @ Wo(f32) + bo -> f32 d_out [B,S,D]
// ---------------------------------------------------------------------------
__global__ __launch_bounds__(256) void out_proj_kernel(
    const unsigned short* __restrict__ Actx, const float* __restrict__ W,
    const float* __restrict__ bias, float* __restrict__ outp)
{
  __shared__ unsigned short As[128 * 40];
  __shared__ unsigned short Bsh[128 * 40];
  const int tid = threadIdx.x, lane = tid & 31, wid = tid >> 5;
  const int wm = wid & 3, wn = wid >> 2;
  const int m0 = blockIdx.y * 128, n0 = blockIdx.x * 128;

  v8f acc[2][4];
  for (int i = 0; i < 2; ++i) for (int j = 0; j < 4; ++j) acc[i][j] = v8f{};

  const int tk8 = (tid & 3) * 8, tmA = tid >> 2;     // A: 2 rows x 8 bf16
  const int tnB = (tid & 31) * 4, kgB = tid >> 5;

  for (int k0 = 0; k0 < Dd; k0 += 32) {
    for (int i = 0; i < 2; ++i) {
      int row = tmA + 64 * i;
      *(uvec4*)(&As[row * 40 + tk8]) =
          *(const uvec4*)(Actx + (size_t)(m0 + row) * Dd + k0 + tk8);
    }
    for (int j = 0; j < 4; ++j) {
      int k = kgB * 4 + j;
      fvec4 wv = *(const fvec4*)(W + (size_t)(k0 + k) * Dd + n0 + tnB);
      for (int c = 0; c < 4; ++c) Bsh[(tnB + c) * 40 + k] = f2bf(wv[c]);
    }
    if (k0 + 32 < Dd) {
      __builtin_prefetch((const void*)(Actx + (size_t)(m0 + tmA) * Dd + k0 + 32 + tk8), 0, 3);
      __builtin_prefetch((const void*)(W + (size_t)(k0 + 32 + kgB * 4) * Dd + n0 + tnB), 0, 3);
    }
    __syncthreads();
    v16bf a0 = frag_a(&As[(wm * 32) * 40], 40, lane);
    v16bf a1 = frag_a(&As[(wm * 32 + 16) * 40], 40, lane);
    for (int nt = 0; nt < 4; ++nt) {
      v16bf bq = frag_b(&Bsh[(wn * 64 + nt * 16) * 40], 40, lane);
      acc[0][nt] = wmma_bf16(a0, bq, acc[0][nt]);
      acc[1][nt] = wmma_bf16(a1, bq, acc[1][nt]);
    }
    __syncthreads();
  }

  const int mb = (lane >> 4) * 8, nn = lane & 15;
  for (int mt = 0; mt < 2; ++mt)
    for (int nt = 0; nt < 4; ++nt)
      for (int r = 0; r < 8; ++r) {
        int mg = m0 + wm * 32 + mt * 16 + mb + r;
        int ng = n0 + wn * 64 + nt * 16 + nn;
        outp[(size_t)mg * Dd + ng] = acc[mt][nt][r] + bias[ng];
      }
}

extern "C" void kernel_launch(void* const* d_in, const int* in_sizes, int n_in,
                              void* d_out, int out_size, void* d_ws, size_t ws_size,
                              hipStream_t stream) {
  (void)in_sizes; (void)n_in; (void)out_size; (void)ws_size;
  const float* Q  = (const float*)d_in[0];
  const float* K  = (const float*)d_in[1];
  const float* V  = (const float*)d_in[2];
  // d_in[3] = causal mask (bool) -- computed analytically, unused
  const float* Wq = (const float*)d_in[4];
  const float* bq = (const float*)d_in[5];
  const float* Wk = (const float*)d_in[6];
  const float* bk = (const float*)d_in[7];
  const float* Wv = (const float*)d_in[8];
  const float* bv = (const float*)d_in[9];
  const float* Wo = (const float*)d_in[10];
  const float* bo = (const float*)d_in[11];
  float* out = (float*)d_out;

  unsigned short* qh  = (unsigned short*)d_ws;               // [B,H,S,64] bf16
  unsigned short* kh  = qh + (size_t)Mrows * Dd;
  unsigned short* vh  = kh + (size_t)Mrows * Dd;
  unsigned short* ctx = vh + (size_t)Mrows * Dd;             // [B,S,D] bf16

  dim3 gg(Dd / 128, Mrows / 128);
  qkv_proj_kernel<<<gg, 256, 0, stream>>>(Q, Wq, bq, qh);
  qkv_proj_kernel<<<gg, 256, 0, stream>>>(K, Wk, bk, kh);
  qkv_proj_kernel<<<gg, 256, 0, stream>>>(V, Wv, bv, vh);
  attn_kernel<<<dim3(Ss / 128, Bb * Hh), 128, 0, stream>>>(qh, kh, vh, ctx);
  out_proj_kernel<<<gg, 256, 0, stream>>>(ctx, Wo, bo, out);
}
